// SocialAgg_31267361915504
// MI455X (gfx1250) — matrix operand order repaired and checked by reading, not compiled, not run on hardware
//
#include <hip/hip_runtime.h>
#include <hip/hip_bf16.h>

#define N_NODES 50000
#define N_EDGES 800000
#define DIM 64

typedef __attribute__((ext_vector_type(16))) __bf16 v16bf;
typedef __attribute__((ext_vector_type(8)))  float  v8f;
typedef __attribute__((ext_vector_type(4)))  float  f4v;
typedef __attribute__((ext_vector_type(4)))  unsigned int v4u;
typedef __attribute__((ext_vector_type(8)))  int    v8i;
typedef __attribute__((ext_vector_type(4)))  int    v4i;

union ABu { v16bf v; unsigned int u[8]; };

#if __has_builtin(__builtin_amdgcn_tensor_load_to_lds)
#define HAVE_TDM 1
#endif

// Native f32->bf16 (RNE) pair pack; lowers to v_cvt_pk_bf16_f32.
static __device__ __forceinline__ unsigned int cvt_bf16x2(float lo, float hi) {
    union { __bf16 h[2]; unsigned int u; } r;
    r.h[0] = (__bf16)lo;
    r.h[1] = (__bf16)hi;
    return r.u;
}

static __device__ __forceinline__ float neg_inf_f() {
    union { unsigned int i; float f; } u; u.i = 0xFF800000u; return u.f;
}

static __device__ __forceinline__ void atomicMaxF(float* addr, float v) {
    if (v >= 0.f) atomicMax((int*)addr, __float_as_int(v));
    else          atomicMin((unsigned int*)addr, __float_as_uint(v));
}

// ---------------------------------------------------------------------------
// Weight swizzle: store W[K][64] (f32 row-major) as bf16 pairs in the WMMA
// B-matrix per-lane layout. Linear index: ((c*4 + t)*32 + lane)*8 + v
//   n = t*16 + lane%16 ; half = lane/16 ; k = c*32 + (v/4)*16 + half*8 + (v%4)*2 (+p)
// ---------------------------------------------------------------------------
__global__ void prep_weights(const float* __restrict__ w1,
                             const float* __restrict__ w2,
                             const float* __restrict__ ww,
                             unsigned int* __restrict__ w1s,
                             unsigned int* __restrict__ w2s,
                             unsigned int* __restrict__ wws) {
    int stride = gridDim.x * blockDim.x;
    for (int i = blockIdx.x * blockDim.x + threadIdx.x; i < 8192; i += stride) {
        const float* W; unsigned int* dst; int idx;
        if (i < 4096)      { W = w1; dst = w1s; idx = i;        }   // K=128 -> 4 chunks
        else if (i < 6144) { W = w2; dst = w2s; idx = i - 4096; }   // K=64  -> 2 chunks
        else               { W = ww; dst = wws; idx = i - 6144; }   // K=64  -> 2 chunks
        int v    = idx & 7;
        int lane = (idx >> 3) & 31;
        int ct   = idx >> 8;          // c*4 + t
        int t    = ct & 3;
        int c    = ct >> 2;
        int n    = t * 16 + (lane & 15);
        int half = lane >> 4;
        int k    = c * 32 + (v >> 2) * 16 + half * 8 + (v & 3) * 2;
        dst[idx] = cvt_bf16x2(W[k * DIM + n], W[(k + 1) * DIM + n]);
    }
}

__global__ void init_buffers(float* __restrict__ mmax, float* __restrict__ ssum,
                             float* __restrict__ agg) {
    int stride = gridDim.x * blockDim.x;
    int i0 = blockIdx.x * blockDim.x + threadIdx.x;
    for (int j = i0; j < N_NODES; j += stride) { mmax[j] = neg_inf_f(); ssum[j] = 0.f; }
    for (int j = i0; j < N_NODES * DIM; j += stride) agg[j] = 0.f;
}

// ---------------------------------------------------------------------------
// Per-edge attention MLP + fused segment-max.  One wave = 16 edges.
// Weights staged in LDS via the Tensor Data Mover (one 24KB 1-row tile),
// fallback to cooperative copy. H1 transpose tile kept as bf16 in LDS.
// ---------------------------------------------------------------------------
__global__ void __launch_bounds__(256) edge_mlp_score(
        const float* __restrict__ uf, const float* __restrict__ hi,
        const unsigned int* __restrict__ w1s, const float* __restrict__ b1,
        const float* __restrict__ b2,
        const float* __restrict__ w3, const float* __restrict__ b3,
        const long long* __restrict__ rows, const long long* __restrict__ cols,
        float* __restrict__ score, float* __restrict__ mmax) {
    __shared__ __align__(16) unsigned int wlds[6144];   // w1s (4096) + w2s (2048)
    __shared__ __align__(16) __bf16 h1[8][16][72];      // per-wave 16x64 (+pad)

#ifdef HAVE_TDM
    // ---- TDM stage: one 6144-dword row from ws (w1s||w2s) into LDS ----
    if (threadIdx.x < 32) {                    // wave 0 issues once; EXEC ignored
        unsigned long long ga = (unsigned long long)(const void*)w1s;
        unsigned int lbase = (unsigned int)(unsigned long long)(const void*)&wlds[0];
        const unsigned int D0 = 6144u;         // tensor_dim0 == tile_dim0 (dwords)
        v4u g0;
        g0[0] = 1u;                                        // count=1 (valid user D#)
        g0[1] = lbase;                                     // lds_addr
        g0[2] = (unsigned int)(ga & 0xFFFFFFFFu);          // global_addr[31:0]
        g0[3] = (unsigned int)((ga >> 32) & 0x01FFFFFFu) | (2u << 30); // [56:32]|type=2
        v8i g1;
        g1[0] = (int)(2u << 16);                           // data_size = 4 bytes
        g1[1] = (int)((D0 & 0xFFFFu) << 16);               // tensor_dim0[15:0]
        g1[2] = (int)(((D0 >> 16) & 0xFFFFu) | (1u << 16)); // dim0[31:16] | dim1=1
        g1[3] = (int)((D0 & 0xFFFFu) << 16);               // dim1 hi(0) | tile_dim0
        g1[4] = 1;                                         // tile_dim1=1, tile_dim2=0
        g1[5] = (int)D0;                                   // tensor_dim0_stride
        g1[6] = 0;
        g1[7] = 0;
        v4i gz = {0, 0, 0, 0};
#if defined(__clang_major__) && (__clang_major__ >= 23)
        v8i gz8 = {0, 0, 0, 0, 0, 0, 0, 0};
        __builtin_amdgcn_tensor_load_to_lds(g0, g1, gz, gz, gz8, 0);
#else
        __builtin_amdgcn_tensor_load_to_lds(g0, g1, gz, gz, 0);
#endif
        __builtin_amdgcn_s_wait_tensorcnt(0);
    }
    __syncthreads();
#else
    for (int i = threadIdx.x; i < 1536; i += 256)
        ((uint4*)wlds)[i] = ((const uint4*)w1s)[i];
    __syncthreads();
#endif

    const int lane = threadIdx.x & 31;
    const int wave = threadIdx.x >> 5;
    const int tile = blockIdx.x * 8 + wave;            // 50000 tiles exactly
    const int m    = lane & 15;                        // edge-in-tile / A row
    const int half = lane >> 4;
    const int e    = tile * 16 + m;
    const long long r = rows[e];
    const long long c = cols[e];
    const float* rowU = uf + r * (long long)DIM;
    const float* rowH = hi + c * (long long)DIM;

    // ---- layer 1: K=128 in 4 chunks of 32, N=64 in 4 tiles of 16 ----
    v8f acc1[4] = {};
    for (int ck = 0; ck < 4; ++ck) {
        const float* src = (ck < 2) ? rowU : rowH;
        const int qb = (ck & 1) * 2;
        f4v f[4];                                      // the lane's 16 floats
        #pragma unroll
        for (int qq = 0; qq < 2; ++qq) {
            const f4v* p = (const f4v*)(src + (qb + qq) * 16 + half * 8);
            f[qq * 2]     = p[0];
            f[qq * 2 + 1] = p[1];
        }
        ABu A;
        #pragma unroll
        for (int v = 0; v < 8; ++v) {                  // A layout pack
            const f4v& x = f[(v >> 2) * 2 + ((v & 3) >> 1)];
            int c0 = (v & 1) * 2;
            A.u[v] = cvt_bf16x2(x[c0], x[c0 + 1]);
        }
        const unsigned int* wblk = wlds + ck * 1024 + lane * 8;
        ABu Bf[4];
        #pragma unroll
        for (int t = 0; t < 4; ++t)
            #pragma unroll
            for (int v = 0; v < 8; ++v) Bf[t].u[v] = wblk[t * 256 + v];
        #pragma unroll
        for (int t = 0; t < 4; ++t)
            acc1[t] = __builtin_amdgcn_wmma_f32_16x16x32_bf16(
                false, A.v, false, Bf[t].v, (short)0, acc1[t], false, false);
    }
    // bias + relu; transpose D-layout -> LDS [edge][dim] as bf16
    #pragma unroll
    for (int t = 0; t < 4; ++t) {
        float bb = b1[t * 16 + m];                     // column n = t*16 + lane%16
        #pragma unroll
        for (int j = 0; j < 8; ++j) {
            float x = acc1[t][j] + bb;
            h1[wave][j + half * 8][t * 16 + m] = (__bf16)(x > 0.f ? x : 0.f);
        }
    }
    asm volatile("s_wait_dscnt 0" ::: "memory");

    // ---- layer 2: K=64 in 2 chunks, N=64 in 4 tiles ----
    v8f acc2[4] = {};
    for (int ck = 0; ck < 2; ++ck) {
        ABu A;
        #pragma unroll
        for (int v = 0; v < 8; ++v) {
            int k = ck * 32 + (v >> 2) * 16 + half * 8 + (v & 3) * 2;
            A.u[v] = *(const unsigned int*)&h1[wave][m][k];   // bf16 pair, k even
        }
        const unsigned int* wblk = wlds + 4096 + ck * 1024 + lane * 8;
        ABu Bf[4];
        #pragma unroll
        for (int t = 0; t < 4; ++t)
            #pragma unroll
            for (int v = 0; v < 8; ++v) Bf[t].u[v] = wblk[t * 256 + v];
        #pragma unroll
        for (int t = 0; t < 4; ++t)
            acc2[t] = __builtin_amdgcn_wmma_f32_16x16x32_bf16(
                false, A.v, false, Bf[t].v, (short)0, acc2[t], false, false);
    }

    // ---- layer 3: score[m] = relu(H2+b2) . w3 + b3, via shfl reduction ----
    float part[8];
    #pragma unroll
    for (int j = 0; j < 8; ++j) part[j] = 0.f;
    #pragma unroll
    for (int t = 0; t < 4; ++t) {
        float bb = b2[t * 16 + m];
        float wv = w3[t * 16 + m];
        #pragma unroll
        for (int j = 0; j < 8; ++j) {
            float x = acc2[t][j] + bb;
            part[j] += (x > 0.f ? x : 0.f) * wv;
        }
    }
    #pragma unroll
    for (int mask = 8; mask; mask >>= 1) {             // stays within 16-lane halves
        #pragma unroll
        for (int j = 0; j < 8; ++j) part[j] += __shfl_xor(part[j], mask, 32);
    }
    if ((lane & 15) == 0) {                            // lanes 0 and 16
        float bb3 = b3[0];
        #pragma unroll
        for (int j = 0; j < 8; ++j) {
            int ee = tile * 16 + half * 8 + j;         // D layout: M = j + 8*half
            float s = part[j] + bb3;
            score[ee] = s;
            atomicMaxF(&mmax[(int)cols[ee]], s);
        }
    }
}

__global__ void edge_exp(const float* __restrict__ score,
                         const long long* __restrict__ cols,
                         const float* __restrict__ mmax,
                         float* __restrict__ expv, float* __restrict__ ssum) {
    int stride = gridDim.x * blockDim.x;
    for (int i = blockIdx.x * blockDim.x + threadIdx.x; i < N_EDGES; i += stride) {
        int c = (int)cols[i];
        float v = __expf(score[i] - mmax[c]);
        expv[i] = v;
        atomicAdd(&ssum[c], v);
    }
}

__global__ void edge_alpha(float* __restrict__ expv,
                           const float* __restrict__ ssum,
                           const long long* __restrict__ cols) {
    int stride = gridDim.x * blockDim.x;
    for (int i = blockIdx.x * blockDim.x + threadIdx.x; i < N_EDGES; i += stride)
        expv[i] = expv[i] / ssum[(int)cols[i]];        // expv becomes alpha
}

__global__ void edge_scatter(const float* __restrict__ alpha,
                             const float* __restrict__ hi,
                             const long long* __restrict__ rows,
                             const long long* __restrict__ cols,
                             float* __restrict__ agg) {
    int stride = gridDim.x * blockDim.x;
    const int total = N_EDGES * DIM;                   // 51.2M, fits in int
    for (int i = blockIdx.x * blockDim.x + threadIdx.x; i < total; i += stride) {
        int e  = i >> 6;
        int dd = i & 63;
        int r  = (int)rows[e];
        int c  = (int)cols[e];
        atomicAdd(&agg[c * DIM + dd], alpha[e] * hi[r * DIM + dd]);
    }
}

// ---------------------------------------------------------------------------
// hs = relu(agg @ w_w + w_b) : one wave per 16-node tile, WMMA bf16.
// ---------------------------------------------------------------------------
__global__ void __launch_bounds__(256) node_out(
        const float* __restrict__ agg, const unsigned int* __restrict__ wws,
        const float* __restrict__ wb, float* __restrict__ out) {
    const int lane = threadIdx.x & 31;
    const int wave = threadIdx.x >> 5;
    const int tile = blockIdx.x * 8 + wave;            // 3125 tiles
    if (tile >= (N_NODES / 16)) return;                // wave-uniform exit
    const int m    = lane & 15;
    const int half = lane >> 4;
    const float* row = agg + (tile * 16 + m) * DIM;

    v8f accs[4] = {};
    for (int ck = 0; ck < 2; ++ck) {
        f4v f[4];
        #pragma unroll
        for (int qq = 0; qq < 2; ++qq) {
            const f4v* p = (const f4v*)(row + (ck * 2 + qq) * 16 + half * 8);
            f[qq * 2]     = p[0];
            f[qq * 2 + 1] = p[1];
        }
        ABu A;
        #pragma unroll
        for (int v = 0; v < 8; ++v) {
            const f4v& x = f[(v >> 2) * 2 + ((v & 3) >> 1)];
            int c0 = (v & 1) * 2;
            A.u[v] = cvt_bf16x2(x[c0], x[c0 + 1]);
        }
        const unsigned int* wblk = wws + (ck * 4) * 256 + lane * 8;
        ABu Bf[4];
        #pragma unroll
        for (int t = 0; t < 4; ++t)
            #pragma unroll
            for (int v = 0; v < 8; ++v) Bf[t].u[v] = wblk[t * 256 + v];
        #pragma unroll
        for (int t = 0; t < 4; ++t)
            accs[t] = __builtin_amdgcn_wmma_f32_16x16x32_bf16(
                false, A.v, false, Bf[t].v, (short)0, accs[t], false, false);
    }
    #pragma unroll
    for (int t = 0; t < 4; ++t) {
        int n = t * 16 + m;
        float bb = wb[n];
        #pragma unroll
        for (int j = 0; j < 8; ++j) {
            int node = tile * 16 + j + 8 * half;       // D layout: M = j + 8*half
            float x = accs[t][j] + bb;
            out[node * DIM + n] = x > 0.f ? x : 0.f;
        }
    }
}

extern "C" void kernel_launch(void* const* d_in, const int* in_sizes, int n_in,
                              void* d_out, int out_size, void* d_ws, size_t ws_size,
                              hipStream_t stream) {
    const float* uf = (const float*)d_in[0];
    const float* hi = (const float*)d_in[1];
    const float* w1 = (const float*)d_in[2];
    const float* b1 = (const float*)d_in[3];
    const float* w2 = (const float*)d_in[4];
    const float* b2 = (const float*)d_in[5];
    const float* w3 = (const float*)d_in[6];
    const float* b3 = (const float*)d_in[7];
    const float* ww = (const float*)d_in[8];
    const float* wb = (const float*)d_in[9];
    const long long* rows = (const long long*)d_in[10];
    const long long* cols = (const long long*)d_in[11];
    float* out = (float*)d_out;

    char* ws = (char*)d_ws;
    unsigned int* w1s = (unsigned int*)(ws);                 // 16 KB  (w2s contiguous)
    unsigned int* w2s = (unsigned int*)(ws + 16384);         //  8 KB
    unsigned int* wws = (unsigned int*)(ws + 24576);         //  8 KB
    float* score = (float*)(ws + 32768);                     // 3.2 MB
    float* expv  = score + N_EDGES;                          // 3.2 MB (becomes alpha)
    float* mmax  = expv + N_EDGES;                           // 200 KB
    float* ssum  = mmax + N_NODES;                           // 200 KB
    float* agg   = ssum + N_NODES;                           // 12.8 MB

    prep_weights<<<32, 256, 0, stream>>>(w1, w2, ww, w1s, w2s, wws);
    init_buffers<<<1024, 256, 0, stream>>>(mmax, ssum, agg);
    edge_mlp_score<<<N_EDGES / 16 / 8, 256, 0, stream>>>(uf, hi, w1s, b1, b2,
                                                         w3, b3, rows, cols, score, mmax);
    edge_exp<<<2048, 256, 0, stream>>>(score, cols, mmax, expv, ssum);
    edge_alpha<<<2048, 256, 0, stream>>>(expv, ssum, cols);
    edge_scatter<<<8192, 256, 0, stream>>>(expv, hi, rows, cols, agg);
    node_out<<<(N_NODES / 16 + 7) / 8, 256, 0, stream>>>(agg, wws, wb, out);
}
